// CustomMultiHeadAttention_16243566313770
// MI455X (gfx1250) — matrix operand order repaired and checked by reading, not compiled, not run on hardware
//
#include <hip/hip_runtime.h>

typedef __attribute__((ext_vector_type(16))) __bf16 v16bf;
typedef __attribute__((ext_vector_type(8)))  float  v8f;

#define T_TOK 8192   // B*S
#define EDIM  512
#define SEQ   1024
#define NB    8
#define NH    8
#define HD    64

union Op16  { v16bf v; uint4 u[2]; __bf16 e[16]; };
union F8    { v8f v; float f[8]; };
union Pack8 { __bf16 e[8]; uint4 u; };

__device__ __forceinline__ v8f wmma_bf16(v16bf a, v16bf b, v8f c) {
  // D = A(16x32 bf16) x B(32x16 bf16) + C(16x16 f32)
  return __builtin_amdgcn_wmma_f32_16x16x32_bf16(false, a, false, b, (short)0, c,
                                                 false, false);
}

__device__ __forceinline__ v8f scale8(v8f x, float s) {
  F8 u; u.v = x;
#pragma unroll
  for (int r = 0; r < 8; ++r) u.f[r] *= s;
  return u.v;
}

// ---------------- f32 -> bf16 conversion (vector x4) ----------------
__global__ void cvt_f32_bf16(const float* __restrict__ s, __bf16* __restrict__ d, int n) {
  int i = 4 * (blockIdx.x * blockDim.x + threadIdx.x);
  if (i < n) {
    float4 v = *(const float4*)(s + i);
    union { __bf16 e[4]; uint2 u; } p;
    p.e[0] = (__bf16)v.x; p.e[1] = (__bf16)v.y;
    p.e[2] = (__bf16)v.z; p.e[3] = (__bf16)v.w;
    *(uint2*)(d + i) = p.u;
  }
}

// ---------------- Q/K/O projection GEMM, feature-major C tile ------------------
// Computes Y[t,f] = sum_e X[t,e]*W[f,e] + bias[f] as D = W(16x32) x X^T(32x16):
// C tile has M=feature, N=token, so consecutive acc elements are feature-
// consecutive -> packed 16B epilogue stores. Wave tile: 64(f) x 32(t).
// F32OUT: 0 = bf16 row-major out, 1 = f32 row-major out.
template <int F32OUT>
__global__ __launch_bounds__(128) void gemm_qk(const __bf16* __restrict__ X,
                                               const __bf16* __restrict__ W,
                                               const float* __restrict__ bias,
                                               void* __restrict__ outp) {
  const int lane = threadIdx.x & 31;
  const int wid  = blockIdx.x * 4 + (threadIdx.x >> 5);
  const int t0   = (wid >> 3) * 32;   // 8192/32 = 256 token tiles
  const int f0   = (wid & 7) * 64;    // 512/64  = 8 feature tiles
  const int ql   = lane & 15;
  const int hf   = lane >> 4;

  const v8f vzero = {0.f,0.f,0.f,0.f,0.f,0.f,0.f,0.f};
  v8f acc[4][2];
#pragma unroll
  for (int i = 0; i < 4; ++i)
#pragma unroll
    for (int j = 0; j < 2; ++j) acc[i][j] = vzero;

  for (int e0 = 0; e0 < EDIM; e0 += 32) {
    Op16 a[4], bm[2];
#pragma unroll
    for (int ft = 0; ft < 4; ++ft) {
      // A 16x32 layout: lanes0-15 row=f, K {0-7,16-23}; lanes16-31 K {8-15,24-31}
      const __bf16* wr = W + (size_t)(f0 + 16 * ft + ql) * EDIM + e0 + 8 * hf;
      a[ft].u[0] = *(const uint4*)wr;
      a[ft].u[1] = *(const uint4*)(wr + 16);
    }
#pragma unroll
    for (int tt = 0; tt < 2; ++tt) {
      // B 32x16 layout: lane%16 = N (=token), lanes0-15 K {0-7,16-23}; lanes16-31 K {8-15,24-31}
      const __bf16* xr = X + (size_t)(t0 + 16 * tt + ql) * EDIM + e0 + 8 * hf;
      bm[tt].u[0] = *(const uint4*)xr;
      bm[tt].u[1] = *(const uint4*)(xr + 16);
    }
#pragma unroll
    for (int ft = 0; ft < 4; ++ft)
#pragma unroll
      for (int tt = 0; tt < 2; ++tt)
        acc[ft][tt] = wmma_bf16(a[ft].v, bm[tt].v, acc[ft][tt]);
  }

#pragma unroll
  for (int ft = 0; ft < 4; ++ft) {
    // bias along M (feature) axis: per-lane 8 consecutive values
    float bb[8];
    const float* bp = bias + f0 + 16 * ft + 8 * hf;
    *(float4*)&bb[0] = *(const float4*)bp;
    *(float4*)&bb[4] = *(const float4*)(bp + 4);
#pragma unroll
    for (int tt = 0; tt < 2; ++tt) {
      F8 c; c.v = acc[ft][tt];
      const size_t t = (size_t)(t0 + 16 * tt + ql);      // C layout: N = token = ql
      const size_t fbase = f0 + 16 * ft + 8 * hf;        // M = r (+8 upper half)
      if (F32OUT) {
        float* o = (float*)outp + t * EDIM + fbase;
        float4 lo = {c.f[0] + bb[0], c.f[1] + bb[1], c.f[2] + bb[2], c.f[3] + bb[3]};
        float4 hi = {c.f[4] + bb[4], c.f[5] + bb[5], c.f[6] + bb[6], c.f[7] + bb[7]};
        *(float4*)o = lo;
        *(float4*)(o + 4) = hi;
      } else {
        Pack8 p;
#pragma unroll
        for (int r = 0; r < 8; ++r) p.e[r] = (__bf16)(c.f[r] + bb[r]);
        *(uint4*)((__bf16*)outp + t * EDIM + fbase) = p.u;
      }
    }
  }
}

// ---------------- V projection GEMM -> transposed-per-head vT[b,h,d,s] ---------
// Token-major C tile (M=token, N=feature): consecutive acc elements are
// s-consecutive in vT -> packed 16B stores. Wave tile: 32(t) x 64(f).
__global__ __launch_bounds__(128) void gemm_v(const __bf16* __restrict__ A,
                                              const __bf16* __restrict__ W,
                                              const float* __restrict__ bias,
                                              __bf16* __restrict__ outp) {
  const int lane = threadIdx.x & 31;
  const int wid  = blockIdx.x * 4 + (threadIdx.x >> 5);
  const int t0   = (wid >> 3) * 32;
  const int f0   = (wid & 7) * 64;
  const int ql   = lane & 15;
  const int hf   = lane >> 4;

  const v8f vzero = {0.f,0.f,0.f,0.f,0.f,0.f,0.f,0.f};
  v8f acc[2][4];
#pragma unroll
  for (int i = 0; i < 2; ++i)
#pragma unroll
    for (int j = 0; j < 4; ++j) acc[i][j] = vzero;

  for (int e0 = 0; e0 < EDIM; e0 += 32) {
    Op16 a[2], bm[4];
#pragma unroll
    for (int mt = 0; mt < 2; ++mt) {
      const __bf16* ar = A + (size_t)(t0 + 16 * mt + ql) * EDIM + e0 + 8 * hf;
      a[mt].u[0] = *(const uint4*)ar;
      a[mt].u[1] = *(const uint4*)(ar + 16);
    }
#pragma unroll
    for (int nt = 0; nt < 4; ++nt) {
      const __bf16* wr = W + (size_t)(f0 + 16 * nt + ql) * EDIM + e0 + 8 * hf;
      bm[nt].u[0] = *(const uint4*)wr;
      bm[nt].u[1] = *(const uint4*)(wr + 16);
    }
#pragma unroll
    for (int mt = 0; mt < 2; ++mt)
#pragma unroll
      for (int nt = 0; nt < 4; ++nt)
        acc[mt][nt] = wmma_bf16(a[mt].v, bm[nt].v, acc[mt][nt]);
  }

#pragma unroll
  for (int nt = 0; nt < 4; ++nt) {
    const int f = f0 + 16 * nt + ql;
    const float bv = bias[f];
    const int h = f0 >> 6;
    const int d = f & 63;
#pragma unroll
    for (int mt = 0; mt < 2; ++mt) {
      F8 c; c.v = acc[mt][nt];
      const int b  = t0 >> 10;
      const int s0 = (t0 & 1023) + 16 * mt + 8 * hf;
      Pack8 p;
#pragma unroll
      for (int r = 0; r < 8; ++r) p.e[r] = (__bf16)(c.f[r] + bv);
      *(uint4*)(outp + ((size_t)(b * NH + h) * HD + d) * SEQ + s0) = p.u;
    }
  }
}

// ---------------- flash attention (transposed-score trick, no LDS) -------------
// Wave owns one (b,h,16-query tile). scores^T = K·Q^T so softmax axis is in-lane.
__global__ __launch_bounds__(128) void attn_flash(const __bf16* __restrict__ qb,
                                                  const __bf16* __restrict__ kb,
                                                  const __bf16* __restrict__ vT,
                                                  const float* __restrict__ adj,
                                                  __bf16* __restrict__ ob) {
  const int lane = threadIdx.x & 31;
  const int wid  = blockIdx.x * 4 + (threadIdx.x >> 5);
  const int qt   = wid & 63;           // 64 q-tiles
  const int bh   = wid >> 6;           // 64 (b,h)
  const int b    = bh >> 3;
  const int h    = bh & 7;
  const int q0   = qt * 16;
  const int ql   = lane & 15;
  const int hf   = lane >> 4;
  const float isd = 0.125f;            // 1/sqrt(64)

  const v8f vzero = {0.f,0.f,0.f,0.f,0.f,0.f,0.f,0.f};

  // Q^T as two B-operands (K-dim = d window of 32)
  Op16 qop[2];
  {
    const __bf16* qr = qb + (size_t)(b * SEQ + q0 + ql) * EDIM + h * HD;
#pragma unroll
    for (int ds = 0; ds < 2; ++ds) {
      const __bf16* p = qr + 32 * ds + 8 * hf;
      qop[ds].u[0] = *(const uint4*)p;
      qop[ds].u[1] = *(const uint4*)(p + 16);
    }
  }

  float m = -__builtin_inff();
  float lsum = 0.f;
  v8f acc[4];
#pragma unroll
  for (int dt = 0; dt < 4; ++dt) acc[dt] = vzero;

  const float* adjr = adj + (size_t)(q0 + ql) * SEQ;

  for (int kc = 0; kc < SEQ; kc += 32) {
    F8 s0, s1; s0.v = vzero; s1.v = vzero;
#pragma unroll
    for (int ds = 0; ds < 2; ++ds) {   // keys kc..kc+15
      Op16 ak;
      const __bf16* kr = kb + (size_t)(b * SEQ + kc + ql) * EDIM + h * HD + 32 * ds + 8 * hf;
      ak.u[0] = *(const uint4*)kr;
      ak.u[1] = *(const uint4*)(kr + 16);
      s0.v = wmma_bf16(ak.v, qop[ds].v, s0.v);
    }
#pragma unroll
    for (int ds = 0; ds < 2; ++ds) {   // keys kc+16..kc+31
      Op16 ak;
      const __bf16* kr = kb + (size_t)(b * SEQ + kc + 16 + ql) * EDIM + h * HD + 32 * ds + 8 * hf;
      ak.u[0] = *(const uint4*)kr;
      ak.u[1] = *(const uint4*)(kr + 16);
      s1.v = wmma_bf16(ak.v, qop[ds].v, s1.v);
    }

    // multiplicative adjacency mask before softmax; element r: key = kc(+16) + 8*hf + r
    const float* ap = adjr + kc + 8 * hf;
    float a0[8], a1[8];
    *(float4*)&a0[0] = *(const float4*)ap;
    *(float4*)&a0[4] = *(const float4*)(ap + 4);
    *(float4*)&a1[0] = *(const float4*)(ap + 16);
    *(float4*)&a1[4] = *(const float4*)(ap + 20);

    float cm = -__builtin_inff();
#pragma unroll
    for (int r = 0; r < 8; ++r) {
      s0.f[r] *= isd * a0[r];
      s1.f[r] *= isd * a1[r];
      cm = fmaxf(cm, fmaxf(s0.f[r], s1.f[r]));
    }
    cm = fmaxf(cm, __shfl_xor(cm, 16));
    const float mnew = fmaxf(m, cm);
    const float sc = __expf(m - mnew);   // first iter: exp(-inf)=0

    float p0[8], p1[8];
    float cs = 0.f;
#pragma unroll
    for (int r = 0; r < 8; ++r) {
      p0[r] = __expf(s0.f[r] - mnew);
      p1[r] = __expf(s1.f[r] - mnew);
      cs += p0[r] + p1[r];
    }
    cs += __shfl_xor(cs, 16);
    lsum = lsum * sc + cs;
    m = mnew;
#pragma unroll
    for (int dt = 0; dt < 4; ++dt) acc[dt] = scale8(acc[dt], sc);

    // P as B-operand (K=32 keys, N=query): swap halves with one shfl_xor(16)
    Op16 bp;
#pragma unroll
    for (int r = 0; r < 8; ++r) {
      float sp0 = __shfl_xor(p0[r], 16);
      float sp1 = __shfl_xor(p1[r], 16);
      bp.e[r]     = (__bf16)(hf ? sp1 : p0[r]);
      bp.e[8 + r] = (__bf16)(hf ? p1[r] : sp0);
    }

    // out^T += V^T(16d x 32key) · P(32key x 16q); A rows contiguous in vT
#pragma unroll
    for (int dt = 0; dt < 4; ++dt) {
      Op16 av;
      const __bf16* vr = vT + ((size_t)bh * HD + 16 * dt + ql) * SEQ + kc + 16 * hf;
      av.u[0] = *(const uint4*)vr;
      av.u[1] = *(const uint4*)(vr + 8);
      acc[dt] = wmma_bf16(av.v, bp.v, acc[dt]);
    }
  }

  const float invl = 1.f / lsum;
  const size_t t = (size_t)b * SEQ + q0 + ql;   // C layout: N = query = ql
#pragma unroll
  for (int dt = 0; dt < 4; ++dt) {
    F8 c; c.v = acc[dt];
    Pack8 p;
#pragma unroll
    for (int r = 0; r < 8; ++r) p.e[r] = (__bf16)(c.f[r] * invl);
    *(uint4*)(ob + t * EDIM + h * HD + 16 * dt + 8 * hf) = p.u;
  }
}

// ---------------- residual add + LayerNorm (f32), also emit bf16 for next layer ---
__global__ __launch_bounds__(128) void add_ln_kernel(const float* __restrict__ xin,
                                                     const float* __restrict__ y,
                                                     const float* __restrict__ g,
                                                     const float* __restrict__ bt,
                                                     float* __restrict__ xout,
                                                     __bf16* __restrict__ xbout) {
  const int t = blockIdx.x;
  const int tid = threadIdx.x;
  __shared__ float sm[4];
  const float* xr = xin + (size_t)t * EDIM;
  const float* yr = y + (size_t)t * EDIM;
  float v[4];
  float s = 0.f;
#pragma unroll
  for (int i = 0; i < 4; ++i) {
    int e = tid + 128 * i;
    v[i] = xr[e] + yr[e];
    s += v[i];
  }
  for (int o = 16; o > 0; o >>= 1) s += __shfl_xor(s, o);
  if ((tid & 31) == 0) sm[tid >> 5] = s;
  __syncthreads();
  const float mu = (sm[0] + sm[1] + sm[2] + sm[3]) * (1.f / 512.f);
  __syncthreads();
  float var = 0.f;
#pragma unroll
  for (int i = 0; i < 4; ++i) { float d = v[i] - mu; var += d * d; }
  for (int o = 16; o > 0; o >>= 1) var += __shfl_xor(var, o);
  if ((tid & 31) == 0) sm[tid >> 5] = var;
  __syncthreads();
  const float rstd = rsqrtf((sm[0] + sm[1] + sm[2] + sm[3]) * (1.f / 512.f) + 1e-5f);
#pragma unroll
  for (int i = 0; i < 4; ++i) {
    int e = tid + 128 * i;
    float o = (v[i] - mu) * rstd * g[e] + bt[e];
    xout[(size_t)t * EDIM + e] = o;
    xbout[(size_t)t * EDIM + e] = (__bf16)o;
  }
}

// --------------------------------------------------------------------------------
extern "C" void kernel_launch(void* const* d_in, const int* in_sizes, int n_in,
                              void* d_out, int out_size, void* d_ws, size_t ws_size,
                              hipStream_t stream) {
  const float* x   = (const float*)d_in[0];
  const float* adj = (const float*)d_in[1];
  const float* Wq  = (const float*)d_in[2];
  const float* bq  = (const float*)d_in[3];
  const float* Wk  = (const float*)d_in[4];
  const float* bk  = (const float*)d_in[5];
  const float* Wv  = (const float*)d_in[6];
  const float* bv  = (const float*)d_in[7];
  const float* Wo  = (const float*)d_in[8];
  const float* bo  = (const float*)d_in[9];
  const float* gm  = (const float*)d_in[10];
  const float* bt  = (const float*)d_in[11];

  char* ws = (char*)d_ws;
  size_t off = 0;
  auto carve = [&](size_t bytes) -> char* {
    char* p = ws + off;
    off += (bytes + 255) & ~(size_t)255;
    return p;
  };
  const size_t TE = (size_t)T_TOK * EDIM;
  __bf16* xb = (__bf16*)carve(TE * 2);
  __bf16* wb[2][4];
  for (int l = 0; l < 2; ++l)
    for (int j = 0; j < 4; ++j) wb[l][j] = (__bf16*)carve((size_t)EDIM * EDIM * 2);
  __bf16* qbuf = (__bf16*)carve(TE * 2);
  __bf16* kbuf = (__bf16*)carve(TE * 2);
  __bf16* vT   = (__bf16*)carve(TE * 2);
  __bf16* ob   = (__bf16*)carve(TE * 2);
  float*  yb   = (float*)carve(TE * 4);
  float*  x1   = (float*)carve(TE * 4);

  // one-time bf16 conversions (all L2-resident)
  cvt_f32_bf16<<<(int)(TE / 4 / 256), 256, 0, stream>>>(x, xb, (int)TE);
  const float* Wsrc[4] = {Wq, Wk, Wv, Wo};
  for (int l = 0; l < 2; ++l)
    for (int j = 0; j < 4; ++j)
      cvt_f32_bf16<<<(EDIM * EDIM / 4) / 256, 256, 0, stream>>>(
          Wsrc[j] + (size_t)l * EDIM * EDIM, wb[l][j], EDIM * EDIM);

  const int GEMM_BLOCKS = (T_TOK / 32) * (EDIM / 64) / 4;  // 512 blocks of 4 waves
  const int ATTN_BLOCKS = (NB * NH * (SEQ / 16)) / 4;      // 1024 blocks of 4 waves
  for (int l = 0; l < 2; ++l) {
    const float* xres = (l == 0) ? x : x1;
    gemm_qk<0><<<GEMM_BLOCKS, 128, 0, stream>>>(xb, wb[l][0], bq + l * EDIM, qbuf);
    gemm_qk<0><<<GEMM_BLOCKS, 128, 0, stream>>>(xb, wb[l][1], bk + l * EDIM, kbuf);
    gemm_v<<<GEMM_BLOCKS, 128, 0, stream>>>(xb, wb[l][2], bv + l * EDIM, vT);
    attn_flash<<<ATTN_BLOCKS, 128, 0, stream>>>(qbuf, kbuf, vT, adj, ob);
    gemm_qk<1><<<GEMM_BLOCKS, 128, 0, stream>>>(ob, wb[l][3], bo + l * EDIM, yb);
    float* xo = (l == 0) ? x1 : (float*)d_out;
    add_ln_kernel<<<T_TOK, 128, 0, stream>>>(xres, yb, gm + l * EDIM, bt + l * EDIM, xo, xb);
  }
}
